// OrthoLinear_9543417331949
// MI455X (gfx1250) — matrix operand order repaired and checked by reading, not compiled, not run on hardware
//
#include <hip/hip_runtime.h>

// W4A16-style fused dequant GEMM + dense residual for gfx1250 (CDNA5, wave32).
// out[M,N] = x[M,K] @ ((nib(packed)-8)*scale[n] + ortho[n,k])^T
// bf16 WMMA (v_wmma_f32_16x16x32_bf16), fp32 accumulation.
// Software-pipelined: next K-stage global loads are prefetched into registers
// while the current stage's WMMAs run; LDS single-buffered.

typedef __attribute__((ext_vector_type(16))) __bf16 v16bf;
typedef __attribute__((ext_vector_type(8)))  float  v8f;

#define TILE_M 128
#define TILE_N 128
#define TILE_K 64
#define LDA 72   /* bf16 elems per LDS row: 64 + 8 pad (144B, 16B-aligned rows) */
#define LDB 72

union FragAB { v16bf v; uint4 u[2]; };
union Pack8  { __bf16 h[8]; uint4 u; };
union Pack4  { __bf16 h[4]; uint2 u; };

__global__ __launch_bounds__(256)
void ortho_linear_w4a16_wmma(const float* __restrict__ x,
                             const int*   __restrict__ packed,
                             const float* __restrict__ scales,
                             const float* __restrict__ ortho,
                             float*       __restrict__ out,
                             int M, int N, int K)
{
    __shared__ __align__(16) unsigned short ldsA[TILE_M * LDA];
    __shared__ __align__(16) unsigned short ldsB[TILE_N * LDB];

    const int tid    = threadIdx.x;
    const int lane   = tid & 31;
    const int wave   = tid >> 5;
    const int waveM  = wave & 1;    // 2 waves along M (64 rows each)
    const int waveN  = wave >> 1;   // 4 waves along N (32 cols each)
    const int l16    = lane & 15;
    const int hiHalf = lane >> 4;

    // M walks blockIdx.x so consecutively-launched blocks share one weight
    // strip (hot in L2) while x streams through L2 once.
    const int m0 = blockIdx.x * TILE_M;
    const int n0 = blockIdx.y * TILE_N;

    v8f acc[4][2];
    const v8f vzero = {0.f, 0.f, 0.f, 0.f, 0.f, 0.f, 0.f, 0.f};
#pragma unroll
    for (int i = 0; i < 4; ++i)
#pragma unroll
        for (int j = 0; j < 2; ++j) acc[i][j] = vzero;

    // A staging: 2048 float4 slots = 128 rows x 16 float4; thread covers 8.
    // B staging: 2 threads per weight row, 32 k-values each.
    const int    bn     = tid >> 1;
    const int    bkh    = (tid & 1) * 32;
    const float  bscale = scales[n0 + bn];
    const int*   prow   = packed + (size_t)(n0 + bn) * (size_t)(K >> 1);
    const float* orow   = ortho  + (size_t)(n0 + bn) * (size_t)K;

    // ---- register prefetch buffers (live across the compute phase) ----
    float4 aReg[8];
    int4   pReg[4];
    float4 oReg[8];

    // prologue: load stage k0 = 0
    {
#pragma unroll
        for (int i = 0; i < 8; ++i) {
            int slot = tid + i * 256;
            int m    = slot >> 4;
            int k4   = slot & 15;
            aReg[i] = *(const float4*)(x + (size_t)(m0 + m) * K + k4 * 4);
        }
        const int4*   pp = (const int4*)(prow + (bkh >> 1));
        const float4* po = (const float4*)(orow + bkh);
#pragma unroll
        for (int j = 0; j < 4; ++j) pReg[j] = pp[j];
#pragma unroll
        for (int j = 0; j < 8; ++j) oReg[j] = po[j];
    }

    for (int k0 = 0; k0 < K; k0 += TILE_K) {
        // ---- drain staged registers into LDS (fp32 -> bf16 / dequant) ----
#pragma unroll
        for (int i = 0; i < 8; ++i) {
            int slot = tid + i * 256;
            int m    = slot >> 4;
            int k4   = slot & 15;
            Pack4 p;
            p.h[0] = (__bf16)aReg[i].x; p.h[1] = (__bf16)aReg[i].y;
            p.h[2] = (__bf16)aReg[i].z; p.h[3] = (__bf16)aReg[i].w;
            *(uint2*)(ldsA + m * LDA + k4 * 4) = p.u;
        }
#pragma unroll
        for (int j4 = 0; j4 < 4; ++j4) {
            int4   pk = pReg[j4];
            float4 o0 = oReg[j4 * 2 + 0];
            float4 o1 = oReg[j4 * 2 + 1];
            Pack8 w;
            int q;
            q = pk.x;
            w.h[0] = (__bf16)(((q        & 15) - 8) * bscale + o0.x);
            w.h[1] = (__bf16)((((q >> 4) & 15) - 8) * bscale + o0.y);
            q = pk.y;
            w.h[2] = (__bf16)(((q        & 15) - 8) * bscale + o0.z);
            w.h[3] = (__bf16)((((q >> 4) & 15) - 8) * bscale + o0.w);
            q = pk.z;
            w.h[4] = (__bf16)(((q        & 15) - 8) * bscale + o1.x);
            w.h[5] = (__bf16)((((q >> 4) & 15) - 8) * bscale + o1.y);
            q = pk.w;
            w.h[6] = (__bf16)(((q        & 15) - 8) * bscale + o1.z);
            w.h[7] = (__bf16)((((q >> 4) & 15) - 8) * bscale + o1.w);
            *(uint4*)(ldsB + bn * LDB + bkh + j4 * 8) = w.u;
        }

        __syncthreads();

        // ---- issue next stage's global loads now; they overlap the WMMAs ----
        const int kn = k0 + TILE_K;
        if (kn < K) {
#pragma unroll
            for (int i = 0; i < 8; ++i) {
                int slot = tid + i * 256;
                int m    = slot >> 4;
                int k4   = slot & 15;
                aReg[i] = *(const float4*)(x + (size_t)(m0 + m) * K + kn + k4 * 4);
            }
            const int4*   pp = (const int4*)(prow + (kn >> 1) + (bkh >> 1));
            const float4* po = (const float4*)(orow + kn + bkh);
#pragma unroll
            for (int j = 0; j < 4; ++j) pReg[j] = pp[j];
#pragma unroll
            for (int j = 0; j < 8; ++j) oReg[j] = po[j];
        }

        // ---- compute: two 16x16x32 bf16 WMMA k-steps per stage ----
#pragma unroll
        for (int kt = 0; kt < 2; ++kt) {
            FragAB a[4], b[2];
            // A fragment: lane<16 -> K[0:8)+K[16:24); lane>=16 -> K[8:16)+K[24:32)
            const int ka0 = kt * 32 + (hiHalf ? 8 : 0);
#pragma unroll
            for (int mi = 0; mi < 4; ++mi) {
                const unsigned short* base =
                    ldsA + (waveM * 64 + mi * 16 + l16) * LDA;
                a[mi].u[0] = *(const uint4*)(base + ka0);
                a[mi].u[1] = *(const uint4*)(base + ka0 + 16);
            }
            // B fragment: lane<16 -> K[0:8)+K[16:24); lane>=16 -> K[8:16)+K[24:32)
            const int kb0 = kt * 32 + (hiHalf ? 8 : 0);
#pragma unroll
            for (int ni = 0; ni < 2; ++ni) {
                const unsigned short* base =
                    ldsB + (waveN * 32 + ni * 16 + l16) * LDB;
                b[ni].u[0] = *(const uint4*)(base + kb0);
                b[ni].u[1] = *(const uint4*)(base + kb0 + 16);
            }
#pragma unroll
            for (int mi = 0; mi < 4; ++mi)
#pragma unroll
                for (int ni = 0; ni < 2; ++ni)
                    acc[mi][ni] = __builtin_amdgcn_wmma_f32_16x16x32_bf16(
                        false, a[mi].v, false, b[ni].v,
                        (short)0, acc[mi][ni], false, false);
        }

        __syncthreads();
    }

    // ---- epilogue: C layout -> flat [M, OUT] fp32 ----
    // C/D layout: VGPR r, lanes 0-15 -> M=r, N=lane; lanes 16-31 -> M=r+8.
#pragma unroll
    for (int mi = 0; mi < 4; ++mi) {
#pragma unroll
        for (int ni = 0; ni < 2; ++ni) {
            const int col  = n0 + waveN * 32 + ni * 16 + l16;
            const int rowb = m0 + waveM * 64 + mi * 16 + hiHalf * 8;
#pragma unroll
            for (int r = 0; r < 8; ++r)
                out[(size_t)(rowb + r) * N + col] = acc[mi][ni][r];
        }
    }
}

extern "C" void kernel_launch(void* const* d_in, const int* in_sizes, int n_in,
                              void* d_out, int out_size, void* d_ws, size_t ws_size,
                              hipStream_t stream) {
    const float* x      = (const float*)d_in[0];   // [B,S,IN] fp32
    const int*   packed = (const int*)d_in[1];     // [OUT, IN/2] int32 (2 nibbles in low byte)
    const float* scales = (const float*)d_in[2];   // [OUT] fp32
    const float* ortho  = (const float*)d_in[3];   // [OUT, IN] fp32 (sparse-with-zeros)
    float*       out    = (float*)d_out;           // [B,S,OUT] fp32

    const int N = in_sizes[2];          // OUT = 4096
    const int K = in_sizes[3] / N;      // IN  = 4096
    const int M = in_sizes[0] / K;      // B*S = 8192

    dim3 grid(M / TILE_M, N / TILE_N);  // 64 x 32 workgroups, M-fastest
    ortho_linear_w4a16_wmma<<<grid, dim3(256), 0, stream>>>(
        x, packed, scales, ortho, out, M, N, K);
}
